// Pyramid_LightNoBN_44547400794766
// MI455X (gfx1250) — matrix operand order repaired and checked by reading, not compiled, not run on hardware
//
#include <hip/hip_runtime.h>
#include <stdint.h>

typedef __attribute__((ext_vector_type(16))) _Float16 v16h;
typedef __attribute__((ext_vector_type(8)))  _Float16 v8h;
typedef __attribute__((ext_vector_type(8)))  float    v8f;
typedef int v4ig __attribute__((vector_size(16)));   // GCC-style int4 (builtin param type)

#define DIV_UP(a,b) (((a)+(b)-1)/(b))

#if defined(__HIP_DEVICE_COMPILE__) && __has_builtin(__builtin_amdgcn_global_load_async_to_lds_b128)
#define ASYNC_FILL 1
#else
#define ASYNC_FILL 0
#endif

#define AS1 __attribute__((address_space(1)))
#define AS3 __attribute__((address_space(3)))

__device__ __forceinline__ void wait_asynccnt0() {
#if defined(__HIP_DEVICE_COMPILE__)
#if __has_builtin(__builtin_amdgcn_s_wait_asynccnt)
    __builtin_amdgcn_s_wait_asynccnt(0);
#else
    asm volatile("s_wait_asynccnt 0x0" ::: "memory");
#endif
#endif
}

// ---------------------------------------------------------------- utilities

__global__ void zero_u32_k(uint32_t* p, size_t nwords) {
    size_t i = (size_t)blockIdx.x * blockDim.x + threadIdx.x;
    if (i < nwords) p[i] = 0u;
}

__global__ void scatter_k(const float* __restrict__ vf, const int* __restrict__ coors,
                          _Float16* __restrict__ x0, uint8_t* __restrict__ mask,
                          int n, int H, int W, int CinStore) {
    int i = blockIdx.x * blockDim.x + threadIdx.x;
    if (i >= n) return;
    int d = coors[i*4+1], h = coors[i*4+2], w = coors[i*4+3];
    int pos = (d*H + h)*W + w;
    x0[(size_t)pos * CinStore] = (_Float16)vf[i];
    mask[pos] = 1;
}

// Repack f32 OIDHW (or OxI shortcut) weights into WMMA A-fragments.
// A layout per CDNA5 ISA 7.12.2 (16-bit A, 16x32):
//   lane 0-15 : M=lane,    halves 0..7 -> K=0..7,  halves 8..15 -> K=16..23
//   lane 16-31: M=lane-16, halves 0..7 -> K=8..15, halves 8..15 -> K=24..31
// Storage: frag[((tap*KB + kb)*MT + mt)*512 + lane*16 + h]
__global__ void repack_k(const float* __restrict__ Wsrc, _Float16* __restrict__ frag,
                         int a /*real in-ch*/, int KB, int MT, int ntaps) {
    int t = blockIdx.x * blockDim.x + threadIdx.x;
    int total = ntaps * KB * MT * 32;
    if (t >= total) return;
    int lane = t & 31;
    int r    = t >> 5;
    int mt   = r % MT; r /= MT;
    int kb   = r % KB;
    int tap  = r / KB;
    int M    = mt*16 + (lane & 15);
    int khi  = (lane >= 16) ? 8 : 0;
    size_t dst = ((size_t)(tap*KB + kb)*MT + mt)*512 + (size_t)lane*16;
    #pragma unroll
    for (int h = 0; h < 16; ++h) {
        int K  = ((h < 8) ? h : (h + 8)) + khi;
        int ci = kb*32 + K;
        float v = (ci < a) ? Wsrc[((size_t)M * a + ci) * ntaps + tap] : 0.0f;
        frag[dst + h] = (_Float16)v;
    }
}

// ------------------------------------------------- implicit-GEMM WMMA conv
// One block = two 16-position tiles (N=32); wave w computes out-ch tile mt=w
// for BOTH position tiles, reusing each A-fragment for two WMMAs.
// The Kpad x 32 activation tile (B) is staged in LDS once per tap — via
// GLOBAL_LOAD_ASYNC_TO_LDS_B128 (ASYNCcnt) when available — then all MT waves
// feed their v_wmma_f32_16x16x32_f16 chains from LDS.
// B per ISA (as A): col N = lane&15; lanes 0-15 K 0..7,16..23; lanes 16-31 K 8..15,24..31.
// C/D per ISA: N = lane&15, VGPR r -> M = r + 8*(lane>=16).
template<int KB, int NTAPS, bool RELU, bool ACC>
__global__ void __launch_bounds__(256) conv_wmma_k(
    const _Float16* __restrict__ actIn, int CinStore,
    const _Float16* __restrict__ frag, int MT,
    _Float16* __restrict__ out, int CoutStore,
    const uint8_t* __restrict__ mask,
    int D, int H, int W)
{
    constexpr int KPAD = KB * 32;
    constexpr int LROW = KPAD + 8;               // +8 halves: stagger LDS banks
    __shared__ _Float16 sB[32 * LROW];

    const int tid  = threadIdx.x;
    const int lane = tid & 31;
    const int hi   = lane >> 4;
    const int nl   = lane & 15;
    const int mt   = tid >> 5;                   // wave index == out-ch tile
    const int row  = tid & 31;                   // fill row (0..31)
    const int g    = tid >> 5;                   // fill group (0..MT-1)
    const int blk  = blockIdx.x;

    // fill-row position & coords (this thread stages row `row`)
    const int posF = blk*32 + row;
    const int wF = posF % W;
    const int hF = (posF / W) % H;
    const int dF = posF / (W * H);

    const _Float16* fragPtr = frag + (size_t)mt * 512;
    const size_t fragStep = (size_t)MT * 512;

    v8f acc0 = {}, acc1 = {};
    for (int tap = 0; tap < NTAPS; ++tap) {
        int dd = 0, dh = 0, dw = 0;
        if (NTAPS == 27) { dd = tap/9 - 1; dh = (tap/3)%3 - 1; dw = tap%3 - 1; }
        const int nd = dF + dd, nh = hF + dh, nw = wF + dw;
        const bool valid = (nd >= 0) & (nd < D) & (nh >= 0) & (nh < H) &
                           (nw >= 0) & (nw < W);
        const _Float16* src = actIn + (size_t)((nd*H + nh)*W + nw) * CinStore;

        __syncthreads();                          // previous tap's readers done
        for (int c = g; c < KB*4; c += MT) {      // 16B chunks of this row
            _Float16* dst = sB + row*LROW + c*8;
#if ASYNC_FILL
            if (valid) {
                __builtin_amdgcn_global_load_async_to_lds_b128(
                    (AS1 v4ig*)(src + c*8), (AS3 v4ig*)dst, 0, 0);
            } else {
                *(v8h*)dst = (v8h){};
            }
#else
            v8h v = {};
            if (valid) v = *(const v8h*)(src + c*8);
            if (RELU) {
                #pragma unroll
                for (int i = 0; i < 8; ++i)
                    v[i] = (v[i] > (_Float16)0) ? v[i] : (_Float16)0;
            }
            *(v8h*)dst = v;
#endif
        }
#if ASYNC_FILL
        wait_asynccnt0();
#endif
        __syncthreads();                          // tile staged

        #pragma unroll
        for (int kb = 0; kb < KB; ++kb) {
            v16h a = *(const v16h*)(fragPtr + (size_t)lane*16);
            fragPtr += fragStep;

            const _Float16* bp0 = sB + nl*LROW        + kb*32 + hi*8;
            const _Float16* bp1 = sB + (16+nl)*LROW   + kb*32 + hi*8;
            v8h x0 = *(const v8h*)bp0, x1 = *(const v8h*)(bp0 + 16);
            v8h y0 = *(const v8h*)bp1, y1 = *(const v8h*)(bp1 + 16);
#if ASYNC_FILL
            if (RELU) {
                #pragma unroll
                for (int i = 0; i < 8; ++i) {
                    x0[i] = (x0[i] > (_Float16)0) ? x0[i] : (_Float16)0;
                    x1[i] = (x1[i] > (_Float16)0) ? x1[i] : (_Float16)0;
                    y0[i] = (y0[i] > (_Float16)0) ? y0[i] : (_Float16)0;
                    y1[i] = (y1[i] > (_Float16)0) ? y1[i] : (_Float16)0;
                }
            }
#endif
            v16h b0, b1;
            #pragma unroll
            for (int i = 0; i < 8; ++i) {
                b0[i] = x0[i]; b0[i+8] = x1[i];
                b1[i] = y0[i]; b1[i+8] = y1[i];
            }
            acc0 = __builtin_amdgcn_wmma_f32_16x16x32_f16(
                       false, a, false, b0, (short)0, acc0, false, false);
            acc1 = __builtin_amdgcn_wmma_f32_16x16x32_f16(
                       false, a, false, b1, (short)0, acc1, false, false);
        }
    }

    // epilogue: mask, optional residual accumulate, vector store (2 tiles)
    const int pos0 = blk*32 + nl;
    const int pos1 = pos0 + 16;
    const float mv0 = mask[pos0] ? 1.0f : 0.0f;
    const float mv1 = mask[pos1] ? 1.0f : 0.0f;
    const size_t ob0 = (size_t)pos0 * CoutStore + (size_t)mt*16 + hi*8;
    const size_t ob1 = (size_t)pos1 * CoutStore + (size_t)mt*16 + hi*8;
    v8h r0, r1;
    if (ACC) {
        v8h o0 = *(const v8h*)(out + ob0);
        v8h o1 = *(const v8h*)(out + ob1);
        #pragma unroll
        for (int r = 0; r < 8; ++r) {
            r0[r] = (_Float16)((float)o0[r] + acc0[r]*mv0);
            r1[r] = (_Float16)((float)o1[r] + acc1[r]*mv1);
        }
    } else {
        #pragma unroll
        for (int r = 0; r < 8; ++r) {
            r0[r] = (_Float16)(acc0[r]*mv0);
            r1[r] = (_Float16)(acc1[r]*mv1);
        }
    }
    *(v8h*)(out + ob0) = r0;
    *(v8h*)(out + ob1) = r1;
}

template<int KB, int NTAPS, bool RELU, bool ACC>
static inline void run_conv(const void* in, int CinStore, const _Float16* frag, int MT,
                            void* out, int CoutStore, const uint8_t* mask,
                            int D, int H, int W, hipStream_t s) {
    int NB = (D*H*W) / 32;          // two 16-pos tiles per block
    conv_wmma_k<KB,NTAPS,RELU,ACC><<<NB, MT*32, 0, s>>>(
        (const _Float16*)in, CinStore, frag, MT,
        (_Float16*)out, CoutStore, mask, D, H, W);
}

// --------------------------------------------- fused ReLU + sparse maxpool
__global__ void pool_k(const _Float16* __restrict__ in, int CinStore, int Creal,
                       _Float16* __restrict__ out, int CoutStore,
                       const uint8_t* __restrict__ mIn, uint8_t* __restrict__ mOut,
                       int Do, int Ho, int Wo)
{
    int t = blockIdx.x * blockDim.x + threadIdx.x;
    int Np = Do*Ho*Wo;
    if (t >= Np * CoutStore) return;
    int ch  = t % CoutStore;
    int pos = t / CoutStore;
    int wo = pos % Wo, ho = (pos/Wo) % Ho, dz = pos / (Wo*Ho);
    int Hi = Ho*2, Wi = Wo*2;
    float acc = 0.0f;     // relu floor: masked negatives clamp to 0, empty -> 0
    int any = 0;
    #pragma unroll
    for (int k = 0; k < 8; ++k) {
        int di = dz*2 + (k>>2), h2 = ho*2 + ((k>>1)&1), wi = wo*2 + (k&1);
        int pi = (di*Hi + h2)*Wi + wi;
        if (mIn[pi]) {
            any = 1;
            if (ch < Creal) {
                float v = (float)in[(size_t)pi*CinStore + ch];
                if (v > acc) acc = v;
            }
        }
    }
    out[(size_t)pos*CoutStore + ch] = (ch < Creal) ? (_Float16)acc : (_Float16)0.0f;
    if (ch == 0) mOut[pos] = (uint8_t)any;
}

// ---------------------------------------------------- depthwise head + bias
__global__ void head_k(const _Float16* __restrict__ in, const float* __restrict__ zW,
                       const float* __restrict__ zb, float* __restrict__ outp)
{
    int t = blockIdx.x * blockDim.x + threadIdx.x;
    if (t >= 128*9) return;
    int c  = t / 9;
    int sp = t % 9;
    float acc = 0.0f;
    #pragma unroll
    for (int kd = 0; kd < 8; ++kd)
        acc += (float)in[(size_t)(kd*9 + sp)*128 + c] * zW[c*8 + kd];
    acc += zb[c];
    outp[c*9 + sp] = (acc > 0.0f) ? acc : 0.0f;
}

// ---------------------------------------------------------------- host side

struct CW { int idx, Cout, a, Kpad, ntaps; size_t off; };

extern "C" void kernel_launch(void* const* d_in, const int* in_sizes, int n_in,
                              void* d_out, int out_size, void* d_ws, size_t ws_size,
                              hipStream_t stream) {
    (void)in_sizes; (void)n_in; (void)out_size; (void)ws_size;
    const float* vf    = (const float*)d_in[0];
    const int*   coors = (const int*)d_in[1];
    const float* zW    = (const float*)d_in[23];
    const float* zb    = (const float*)d_in[24];
    uint8_t* ws = (uint8_t*)d_ws;

    const size_t NP0 = 294912;                  // 128*48*48
    const size_t BUFSZ   = NP0 * 64 * 2;        // 37,748,736 B
    const size_t BUF0    = 0;
    const size_t BUF1    = BUFSZ;
    const size_t BUF2    = 2*BUFSZ;
    const size_t BUF2SZ  = NP0 * 32 * 2;        // 18,874,368 B
    const size_t FRAG    = BUF2 + BUF2SZ;

    // weight/fragment descriptors (d_in index, Cout, a, Kpad, ntaps)
    CW cw[20] = {
        { 3, 64,  1, 32, 1}, { 4, 64,  1, 32,27}, { 5, 64, 64, 64,27},
        { 6, 64, 64, 64,27}, { 7, 64, 64, 64,27},
        { 8, 80, 64, 64, 1}, { 9, 80, 64, 64,27}, {10, 80, 80, 96,27},
        {11, 80, 80, 96,27}, {12, 80, 80, 96,27},
        {13, 96, 80, 96, 1}, {14, 96, 80, 96,27}, {15, 96, 96, 96,27},
        {16, 96, 96, 96,27}, {17, 96, 96, 96,27},
        {18,128, 96, 96, 1}, {19,128, 96, 96,27}, {20,128,128,128,27},
        {21,128,128,128,27}, {22,128,128,128,27},
    };
    size_t run = 0;
    for (int i = 0; i < 20; ++i) {
        cw[i].off = run;
        run += (size_t)cw[i].ntaps * (cw[i].Kpad/32) * (cw[i].Cout/16) * 512;
    }
    const size_t M0 = FRAG + run*2;
    const size_t M1 = M0 + NP0;          // masks (u8)
    const size_t M2 = M1 + 36864;
    const size_t M3 = M2 + 4608;
    const size_t M4 = M3 + 576;

    _Float16* FB = (_Float16*)(ws + FRAG);

    // 1. zero-init X0 (BUF2) and mask0
    {
        size_t nw = BUF2SZ / 4;
        zero_u32_k<<<(unsigned)DIV_UP(nw,256), 256, 0, stream>>>((uint32_t*)(ws + BUF2), nw);
        nw = NP0 / 4;
        zero_u32_k<<<(unsigned)DIV_UP(nw,256), 256, 0, stream>>>((uint32_t*)(ws + M0), nw);
    }
    // 2. repack all weights to fp16 WMMA fragments
    for (int i = 0; i < 20; ++i) {
        int total = cw[i].ntaps * (cw[i].Kpad/32) * (cw[i].Cout/16) * 32;
        repack_k<<<DIV_UP(total,256), 256, 0, stream>>>(
            (const float*)d_in[cw[i].idx], FB + cw[i].off,
            cw[i].a, cw[i].Kpad/32, cw[i].Cout/16, cw[i].ntaps);
    }
    // 3. scatter voxels
    scatter_k<<<DIV_UP(30000,256), 256, 0, stream>>>(
        vf, coors, (_Float16*)(ws + BUF2), ws + M0, 30000, 48, 48, 32);

    auto pool = [&](size_t inOff, int CinStore, int Creal, size_t outOff, int CoutStore,
                    size_t mInOff, size_t mOutOff, int Do, int Ho, int Wo) {
        int total = Do*Ho*Wo*CoutStore;
        pool_k<<<DIV_UP(total,256), 256, 0, stream>>>(
            (const _Float16*)(ws + inOff), CinStore, Creal,
            (_Float16*)(ws + outOff), CoutStore,
            ws + mInOff, ws + mOutOff, Do, Ho, Wo);
    };

    // block 0 @ 128x48x48  (Cin store 32, Cout 64, MT=4)
    run_conv<1, 1,false,false>(ws+BUF2, 32, FB+cw[0].off, 4, ws+BUF0, 64, ws+M0, 128,48,48, stream);
    run_conv<1,27,true ,false>(ws+BUF2, 32, FB+cw[1].off, 4, ws+BUF1, 64, ws+M0, 128,48,48, stream);
    run_conv<2,27,true ,true >(ws+BUF1, 64, FB+cw[2].off, 4, ws+BUF0, 64, ws+M0, 128,48,48, stream);
    run_conv<2,27,true ,false>(ws+BUF0, 64, FB+cw[3].off, 4, ws+BUF1, 64, ws+M0, 128,48,48, stream);
    run_conv<2,27,true ,true >(ws+BUF1, 64, FB+cw[4].off, 4, ws+BUF0, 64, ws+M0, 128,48,48, stream);
    pool(BUF0, 64, 64, BUF2, 64, M0, M1, 64,24,24);

    // block 1 @ 64x24x24 (64 -> 80, stored padded to 96, MT=5)
    run_conv<2, 1,false,false>(ws+BUF2, 64, FB+cw[5].off, 5, ws+BUF0, 96, ws+M1, 64,24,24, stream);
    run_conv<2,27,true ,false>(ws+BUF2, 64, FB+cw[6].off, 5, ws+BUF1, 96, ws+M1, 64,24,24, stream);
    run_conv<3,27,true ,true >(ws+BUF1, 96, FB+cw[7].off, 5, ws+BUF0, 96, ws+M1, 64,24,24, stream);
    run_conv<3,27,true ,false>(ws+BUF0, 96, FB+cw[8].off, 5, ws+BUF1, 96, ws+M1, 64,24,24, stream);
    run_conv<3,27,true ,true >(ws+BUF1, 96, FB+cw[9].off, 5, ws+BUF0, 96, ws+M1, 64,24,24, stream);
    pool(BUF0, 96, 80, BUF2, 96, M1, M2, 32,12,12);

    // block 2 @ 32x12x12 (80 -> 96, MT=6)
    run_conv<3, 1,false,false>(ws+BUF2, 96, FB+cw[10].off, 6, ws+BUF0, 96, ws+M2, 32,12,12, stream);
    run_conv<3,27,true ,false>(ws+BUF2, 96, FB+cw[11].off, 6, ws+BUF1, 96, ws+M2, 32,12,12, stream);
    run_conv<3,27,true ,true >(ws+BUF1, 96, FB+cw[12].off, 6, ws+BUF0, 96, ws+M2, 32,12,12, stream);
    run_conv<3,27,true ,false>(ws+BUF0, 96, FB+cw[13].off, 6, ws+BUF1, 96, ws+M2, 32,12,12, stream);
    run_conv<3,27,true ,true >(ws+BUF1, 96, FB+cw[14].off, 6, ws+BUF0, 96, ws+M2, 32,12,12, stream);
    pool(BUF0, 96, 96, BUF2, 96, M2, M3, 16,6,6);

    // block 3 @ 16x6x6 (96 -> 128, MT=8)
    run_conv<3, 1,false,false>(ws+BUF2, 96, FB+cw[15].off, 8, ws+BUF0,128, ws+M3, 16,6,6, stream);
    run_conv<3,27,true ,false>(ws+BUF2, 96, FB+cw[16].off, 8, ws+BUF1,128, ws+M3, 16,6,6, stream);
    run_conv<4,27,true ,true >(ws+BUF1,128, FB+cw[17].off, 8, ws+BUF0,128, ws+M3, 16,6,6, stream);
    run_conv<4,27,true ,false>(ws+BUF0,128, FB+cw[18].off, 8, ws+BUF1,128, ws+M3, 16,6,6, stream);
    run_conv<4,27,true ,true >(ws+BUF1,128, FB+cw[19].off, 8, ws+BUF0,128, ws+M3, 16,6,6, stream);
    pool(BUF0,128,128, BUF2, 128, M3, M4, 8,3,3);

    // head: depthwise conv over D (k=8) + bias + relu -> f32 output
    head_k<<<DIV_UP(128*9,128), 128, 0, stream>>>(
        (const _Float16*)(ws + BUF2), zW, zb, (float*)d_out);
}
